// LLMAdapterAttention_35476429865277
// MI455X (gfx1250) — matrix operand order repaired and checked by reading, not compiled, not run on hardware
//
#include <hip/hip_runtime.h>
#include <hip/hip_bf16.h>
#include <stdint.h>

// ---------------------------------------------------------------------------
// Problem constants (from reference)
// ---------------------------------------------------------------------------
#define B_   4
#define L_   2048
#define DM   2048
#define NH   16
#define HD   128
#define MROWS (B_ * L_)          // 8192 token rows

typedef __attribute__((ext_vector_type(16))) __bf16 v16bf;
typedef __attribute__((ext_vector_type(8)))  __bf16 v8bf;
typedef __attribute__((ext_vector_type(8)))  float  v8f;

union BF16x16 { v16bf v; v8bf h[2]; };

__device__ inline __bf16 f2b(float f) {
  // round-to-nearest-even fp32 -> bf16
  unsigned u = __float_as_uint(f);
  unsigned r = u + 0x7FFFu + ((u >> 16) & 1u);
  unsigned short s = (unsigned short)(r >> 16);
  __bf16 b;
  __builtin_memcpy(&b, &s, 2);
  return b;
}

__device__ inline v8f wmma_bf16(v16bf a, v16bf b, v8f c) {
  // D(f32 16x16) = A(bf16 16x32) * B(bf16 32x16) + C
  return __builtin_amdgcn_wmma_f32_16x16x32_bf16(
      /*neg_a=*/false, a, /*neg_b=*/false, b,
      /*c_mod=*/(short)0, c, /*reuse_a=*/false, /*reuse_b=*/false);
}

// Async global->LDS copy, 16 bytes per lane (tracked by ASYNCcnt).
// Generic pointers to LDS carry the LDS byte address in their low 32 bits
// (ISA aperture mapping truncates addr[31:0]), so the VDST operand is the
// truncated shared-memory pointer.
__device__ inline void async_cp16(void* lds_ptr, const void* gptr) {
  unsigned lds_addr = (unsigned)(uintptr_t)lds_ptr;
  asm volatile("global_load_async_to_lds_b128 %0, %1, off"
               :: "v"(lds_addr), "v"(gptr)
               : "memory");
}

// ---------------------------------------------------------------------------
// fp32 -> bf16 conversion (grid-stride)
// ---------------------------------------------------------------------------
__global__ void cvt_f32_bf16(const float* __restrict__ in,
                             __bf16* __restrict__ out, int n) {
  int i = blockIdx.x * blockDim.x + threadIdx.x;
  int stride = gridDim.x * blockDim.x;
  for (; i < n; i += stride) out[i] = f2b(in[i]);
}

// ---------------------------------------------------------------------------
// GEMM-NT: C[M,N] = A[M,K] * Bm[N,K]^T   (A,Bm bf16 row-major, C fp32)
// Block = 256 threads = 8 waves; tile 128x128, K-step 32.
// Wave grid 4(M) x 2(N); each wave computes 32x64 via 2x4 WMMA 16x16x32.
// Double-buffered LDS filled by GLOBAL_LOAD_ASYNC_TO_LDS_B128; one-tile-deep
// pipeline via in-order ASYNCcnt (s_wait_asynccnt 4 = previous tile done).
// ---------------------------------------------------------------------------
__global__ __launch_bounds__(256, 2)
void gemm_nt_bf16(const __bf16* __restrict__ A, const __bf16* __restrict__ Bm,
                  float* __restrict__ C, int M, int N, int K) {
  __shared__ __bf16 As[2][128][40];   // +8 bf16 pad per row (rows 16B aligned)
  __shared__ __bf16 Bs[2][128][40];

  const int tid   = threadIdx.x;
  const int lane  = tid & 31;
  const int wave  = tid >> 5;
  const int wm    = wave & 3;      // 0..3 (M)
  const int wn    = wave >> 2;     // 0..1 (N)
  const int ln15  = lane & 15;
  const int khalf = lane >> 4;

  const size_t row0 = (size_t)blockIdx.y * 128;
  const size_t col0 = (size_t)blockIdx.x * 128;

  v8f acc[2][4] = {};

  const int lr = tid >> 2;         // 0..63
  const int lc = (tid & 3) * 8;    // 0,8,16,24

  // issue one 128x32 A-tile + 128x32 B-tile (4 async b128 ops per thread)
  auto issue_tile = [&](int buf, int k0) {
    async_cp16(&As[buf][lr][lc],      &A[(row0 + lr) * K + k0 + lc]);
    async_cp16(&As[buf][lr + 64][lc], &A[(row0 + lr + 64) * K + k0 + lc]);
    async_cp16(&Bs[buf][lr][lc],      &Bm[(col0 + lr) * K + k0 + lc]);
    async_cp16(&Bs[buf][lr + 64][lc], &Bm[(col0 + lr + 64) * K + k0 + lc]);
  };

  issue_tile(0, 0);                // prologue: tile 0 in flight
  int cur = 0;

  for (int k0 = 0; k0 < K; k0 += 32) {
    if (k0 + 32 < K) {
      issue_tile(cur ^ 1, k0 + 32);              // overlap next tile
      asm volatile("s_wait_asynccnt 4" ::: "memory");  // current tile landed
      if (k0 + 64 < K)                            // L2 warm-up, tile after next
        __builtin_prefetch(&A[(row0 + lr) * K + k0 + 64 + lc], 0, 1);
    } else {
      asm volatile("s_wait_asynccnt 0" ::: "memory");
    }
    __syncthreads();                              // all waves' copies visible

    // ---- A fragments: 16x32, lane=row (l&15), khalf selects K group ----
    BF16x16 af[2], bfg[4];
    for (int mf = 0; mf < 2; ++mf) {
      const int r = wm * 32 + mf * 16 + ln15;
      af[mf].h[0] = *(const v8bf*)&As[cur][r][khalf * 8];      // K 0..7 / 8..15
      af[mf].h[1] = *(const v8bf*)&As[cur][r][16 + khalf * 8]; // K 16..23 / 24..31
    }
    // ---- B fragments: 32x16, lane=col, khalf selects K half ----
    for (int nf = 0; nf < 4; ++nf) {
      const int r = wn * 64 + nf * 16 + ln15;
      bfg[nf].h[0] = *(const v8bf*)&Bs[cur][r][khalf * 8];
      bfg[nf].h[1] = *(const v8bf*)&Bs[cur][r][16 + khalf * 8];
    }
    for (int mf = 0; mf < 2; ++mf)
      for (int nf = 0; nf < 4; ++nf)
        acc[mf][nf] = wmma_bf16(af[mf].v, bfg[nf].v, acc[mf][nf]);

    __syncthreads();                              // all reads of cur done
    cur ^= 1;
  }

  // ---- epilogue: C layout VGPR r -> row r (+8 if lane>=16), lane -> col ----
  for (int mf = 0; mf < 2; ++mf) {
    const size_t rb = row0 + wm * 32 + mf * 16 + 8 * khalf;
    for (int nf = 0; nf < 4; ++nf) {
      const size_t cb = col0 + wn * 64 + nf * 16 + ln15;
      for (int r = 0; r < 8; ++r)
        C[(rb + r) * (size_t)N + cb] = acc[mf][nf][r];
    }
  }
}

// ---------------------------------------------------------------------------
// Fused per-head RMSNorm + RoPE.
// In : P fp32 [B, L, NH, HD]  (projection output)
// Out: bf16  [B, NH, L, HD]   (head-major, ready for attention)
// One wave per (b,l,h) row; 4 elems/lane; rotate-half pairs (d, d+64) sit in
// partner lanes (lane ^ 16), handled by one shuffle.
// ---------------------------------------------------------------------------
__global__ __launch_bounds__(256)
void rmsnorm_rope(const float* __restrict__ P, const float* __restrict__ gamma,
                  const float* __restrict__ cs, const float* __restrict__ sn,
                  __bf16* __restrict__ out) {
  const int lane = threadIdx.x & 31;
  const int wave = threadIdx.x >> 5;
  const size_t row = (size_t)blockIdx.x * 8 + wave;   // over B*L*NH
  const int    h  = (int)(row % NH);
  const size_t bl = row / NH;                         // b*L + l
  const float* x  = P + row * HD;
  const float* cp = cs + bl * HD;
  const float* sp = sn + bl * HD;
  const int d0 = lane * 4;

  float xv[4], ss = 0.f;
  for (int i = 0; i < 4; ++i) { xv[i] = x[d0 + i]; ss += xv[i] * xv[i]; }
  for (int off = 16; off >= 1; off >>= 1) ss += __shfl_xor(ss, off, 32);
  const float rinv = rsqrtf(ss * (1.0f / HD) + 1e-6f);

  float xn[4];
  for (int i = 0; i < 4; ++i) xn[i] = xv[i] * rinv * gamma[d0 + i];

  const float sign = (lane < 16) ? -1.f : 1.f;        // rot = [-x_hi, x_lo]
  const int    l = (int)(bl % L_);
  const size_t b = bl / L_;
  __bf16* o = out + (((b * NH + h) * (size_t)L_) + l) * HD;
  for (int i = 0; i < 4; ++i) {
    const float other = __shfl_xor(xn[i], 16, 32);    // xn[d ^ 64]
    o[d0 + i] = f2b(xn[i] * cp[d0 + i] + sign * other * sp[d0 + i]);
  }
}

// ---------------------------------------------------------------------------
// V: fp32 [B, L, NH, HD] -> bf16 transposed [B, NH, HD, L]
// (so P*V B-fragments become contiguous 16B loads)
// ---------------------------------------------------------------------------
__global__ void v_to_bf16_t(const float* __restrict__ P,
                            __bf16* __restrict__ Vt) {
  size_t i = (size_t)blockIdx.x * blockDim.x + threadIdx.x;
  const size_t stride = (size_t)gridDim.x * blockDim.x;
  const size_t n = (size_t)B_ * L_ * NH * HD;
  for (; i < n; i += stride) {
    const int d = (int)(i & (HD - 1));
    const int h = (int)((i >> 7) & (NH - 1));
    const int l = (int)((i >> 11) & (L_ - 1));
    const int b = (int)(i >> 22);
    Vt[(((size_t)b * NH + h) * HD + d) * L_ + l] = f2b(P[i]);
  }
}

// ---------------------------------------------------------------------------
// Flash attention (non-causal). Grid (L/64, NH, B), block 128 (4 waves).
// Each wave owns 16 query rows; iterates keys in chunks of 32.
//   S tile  : 8x WMMA bf16 (Q 16x32-frag x K^T 32x16-frag) over d=128
//   softmax : online, row stats in C-fragment layout, 16-lane xor reductions
//   P       : staged through wave-private LDS (C-layout -> A-layout transpose)
//   P*V     : 8x WMMA bf16 accumulating out[16,128]
// No block barriers: each wave is fully independent (s_wait_dscnt only).
// ---------------------------------------------------------------------------
__global__ __launch_bounds__(128)
void flash_attn(const __bf16* __restrict__ Q, const __bf16* __restrict__ Kk,
                const __bf16* __restrict__ Vt, __bf16* __restrict__ O) {
  __shared__ __bf16 Ps[4][16][40];   // per-wave 16x32 P tile (+pad)

  const int lane  = threadIdx.x & 31;
  const int wave  = threadIdx.x >> 5;
  const int ln15  = lane & 15;
  const int khalf = lane >> 4;
  const int h = blockIdx.y, b = blockIdx.z;
  const int q0 = blockIdx.x * 64 + wave * 16;

  const size_t hb = (size_t)b * NH + h;
  const __bf16* Qh = Q  + hb * L_ * HD;
  const __bf16* Kh = Kk + hb * L_ * HD;
  const __bf16* Vh = Vt + hb * HD * L_;

  // Q fragments: 4 chunks of d (16x32 each), resident for the whole loop.
  BF16x16 qf[4];
  {
    const __bf16* qr = Qh + (size_t)(q0 + ln15) * HD;
    for (int g = 0; g < 4; ++g) {
      qf[g].h[0] = *(const v8bf*)(qr + 32 * g + khalf * 8);
      qf[g].h[1] = *(const v8bf*)(qr + 32 * g + 16 + khalf * 8);
    }
  }

  v8f oacc[8] = {};
  float m_i[8], l_i[8];
  for (int r = 0; r < 8; ++r) { m_i[r] = -3.0e38f; l_i[r] = 0.f; }
  const float SCALE = 0.088388347648318447f;   // 1/sqrt(128)

  for (int j0 = 0; j0 < L_; j0 += 32) {
    if (j0 + 32 < L_)   // warm L2/WGP$ for next key chunk
      __builtin_prefetch(Kh + (size_t)(j0 + 32 + ln15) * HD, 0, 1);

    // ---- S = Q * K^T for keys [j0, j0+32) : two 16x16 tiles ----
    v8f s0 = {}, s1 = {};
    for (int g = 0; g < 4; ++g) {
      BF16x16 kf0, kf1;
      const __bf16* kr0 = Kh + (size_t)(j0 + ln15) * HD + 32 * g + khalf * 8;
      const __bf16* kr1 = Kh + (size_t)(j0 + 16 + ln15) * HD + 32 * g + khalf * 8;
      kf0.h[0] = *(const v8bf*)(kr0); kf0.h[1] = *(const v8bf*)(kr0 + 16);
      kf1.h[0] = *(const v8bf*)(kr1); kf1.h[1] = *(const v8bf*)(kr1 + 16);
      s0 = wmma_bf16(qf[g].v, kf0.v, s0);
      s1 = wmma_bf16(qf[g].v, kf1.v, s1);
    }

    // ---- online softmax, per accumulator row r ----
    for (int r = 0; r < 8; ++r) {
      const float a0 = s0[r] * SCALE, a1 = s1[r] * SCALE;
      float rm = fmaxf(a0, a1);
      for (int off = 1; off < 16; off <<= 1) rm = fmaxf(rm, __shfl_xor(rm, off, 32));
      const float mnew  = fmaxf(m_i[r], rm);
      const float alpha = __expf(m_i[r] - mnew);
      m_i[r] = mnew;
      const float p0 = __expf(a0 - mnew);
      const float p1 = __expf(a1 - mnew);
      l_i[r] = l_i[r] * alpha + p0 + p1;
      for (int g = 0; g < 8; ++g) oacc[g][r] *= alpha;
      const int prow = r + 8 * khalf;            // C-layout row
      Ps[wave][prow][ln15]      = f2b(p0);
      Ps[wave][prow][16 + ln15] = f2b(p1);
    }
    asm volatile("s_wait_dscnt 0" ::: "memory");  // LDS C->A layout handoff

    // ---- reload P as a 16x32 A-fragment, accumulate P * V ----
    BF16x16 pf;
    pf.h[0] = *(const v8bf*)&Ps[wave][ln15][khalf * 8];
    pf.h[1] = *(const v8bf*)&Ps[wave][ln15][16 + khalf * 8];
    for (int g = 0; g < 8; ++g) {
      BF16x16 vf;
      const __bf16* vc = Vh + (size_t)(16 * g + ln15) * L_ + j0 + khalf * 8;
      vf.h[0] = *(const v8bf*)vc; vf.h[1] = *(const v8bf*)(vc + 16);
      oacc[g] = wmma_bf16(pf.v, vf.v, oacc[g]);
    }
  }

  // ---- finalize: divide by row sums, store bf16 [B, L, NH*HD] ----
  float rs[8];
  for (int r = 0; r < 8; ++r) {
    float s = l_i[r];
    for (int off = 1; off < 16; off <<= 1) s += __shfl_xor(s, off, 32);
    rs[r] = 1.0f / s;
  }
  for (int g = 0; g < 8; ++g)
    for (int r = 0; r < 8; ++r) {
      const size_t orow = (size_t)b * L_ + q0 + r + 8 * khalf;
      O[orow * DM + (size_t)h * HD + 16 * g + ln15] = f2b(oacc[g][r] * rs[r]);
    }
}

// ---------------------------------------------------------------------------
// Host orchestration (all launches on `stream`; graph-capture safe)
// ---------------------------------------------------------------------------
extern "C" void kernel_launch(void* const* d_in, const int* in_sizes, int n_in,
                              void* d_out, int out_size, void* d_ws, size_t ws_size,
                              hipStream_t stream) {
  const float* x    = (const float*)d_in[0];
  const float* cosq = (const float*)d_in[1];
  const float* sinq = (const float*)d_in[2];
  const float* cosk = (const float*)d_in[3];
  const float* sink = (const float*)d_in[4];
  const float* Wq   = (const float*)d_in[5];
  const float* Wk   = (const float*)d_in[6];
  const float* Wv   = (const float*)d_in[7];
  const float* Wo   = (const float*)d_in[8];
  const float* qg   = (const float*)d_in[9];
  const float* kg   = (const float*)d_in[10];
  float* out = (float*)d_out;

  char* ws = (char*)d_ws;
  const size_t MB = 1024 * 1024;
  __bf16* xb  = (__bf16*)(ws);              // 32 MB  x  bf16 [B,L,DM]
  __bf16* wqb = (__bf16*)(ws + 32 * MB);    //  8 MB
  __bf16* wkb = (__bf16*)(ws + 40 * MB);    //  8 MB
  __bf16* wvb = (__bf16*)(ws + 48 * MB);    //  8 MB
  __bf16* wob = (__bf16*)(ws + 56 * MB);    //  8 MB
  float*  pf  = (float*) (ws + 64 * MB);    // 64 MB  reused fp32 projection
  __bf16* qb  = (__bf16*)(ws + 128 * MB);   // 32 MB  Q bf16 [B,NH,L,HD]
  __bf16* kb  = (__bf16*)(ws + 160 * MB);   // 32 MB  K bf16 [B,NH,L,HD]
  __bf16* vt  = (__bf16*)(ws + 192 * MB);   // 32 MB  V bf16 [B,NH,HD,L]
  __bf16* ao  = (__bf16*)(ws + 224 * MB);   // 32 MB  attn out bf16 [B,L,DM]

  const int NX = B_ * L_ * DM;   // 16.7M
  const int NW = DM * DM;        // 4.2M

  // 1) precision cast of activations + weights
  cvt_f32_bf16<<<4096, 256, 0, stream>>>(x,  xb,  NX);
  cvt_f32_bf16<<<2048, 256, 0, stream>>>(Wq, wqb, NW);
  cvt_f32_bf16<<<2048, 256, 0, stream>>>(Wk, wkb, NW);
  cvt_f32_bf16<<<2048, 256, 0, stream>>>(Wv, wvb, NW);
  cvt_f32_bf16<<<2048, 256, 0, stream>>>(Wo, wob, NW);

  const dim3 gGemm(DM / 128, MROWS / 128);   // (16, 64)
  const dim3 bGemm(256);

  // 2) Q projection -> RMSNorm+RoPE
  gemm_nt_bf16<<<gGemm, bGemm, 0, stream>>>(xb, wqb, pf, MROWS, DM, DM);
  rmsnorm_rope<<<(B_ * L_ * NH) / 8, 256, 0, stream>>>(pf, qg, cosq, sinq, qb);

  // 3) K projection -> RMSNorm+RoPE
  gemm_nt_bf16<<<gGemm, bGemm, 0, stream>>>(xb, wkb, pf, MROWS, DM, DM);
  rmsnorm_rope<<<(B_ * L_ * NH) / 8, 256, 0, stream>>>(pf, kg, cosk, sink, kb);

  // 4) V projection -> bf16 transpose
  gemm_nt_bf16<<<gGemm, bGemm, 0, stream>>>(xb, wvb, pf, MROWS, DM, DM);
  v_to_bf16_t<<<8192, 256, 0, stream>>>(pf, vt);

  // 5) attention
  flash_attn<<<dim3(L_ / 64, NH, B_), 128, 0, stream>>>(qb, kb, vt, ao);

  // 6) output projection -> fp32 d_out
  gemm_nt_bf16<<<gGemm, bGemm, 0, stream>>>(ao, wob, out, MROWS, DM, DM);
}
